// SingleHeadAttention_24257975288111
// MI455X (gfx1250) — matrix operand order repaired and checked by reading, not compiled, not run on hardware
//
#include <hip/hip_runtime.h>

typedef __attribute__((ext_vector_type(16))) __bf16 v16bf;
typedef __attribute__((ext_vector_type(8)))  float  v8f;

#define E_DIM 1024
#define A_DIM 1024
#define T_DIM 2048
#define B_DIM 4
#define M_DIM (B_DIM * T_DIM)

__device__ __forceinline__ unsigned short f32_to_bf16(float f) {
  unsigned int u = __float_as_uint(f);
  u += 0x7fffu + ((u >> 16) & 1u);   // round-to-nearest-even
  return (unsigned short)(u >> 16);
}

// Low 32 bits of a flat shared-aperture address == LDS byte offset.
__device__ __forceinline__ unsigned lds_off(const void* p) {
  return (unsigned)(size_t)p;
}

// CDNA5 async copy: 16B global -> LDS per lane, tracked by ASYNCcnt.
__device__ __forceinline__ void async_b128(unsigned lds, const unsigned short* g) {
  asm volatile("global_load_async_to_lds_b128 %0, %1, off"
               :: "v"(lds), "v"(g) : "memory");
}
__device__ __forceinline__ void wait_async0() {
  asm volatile("s_wait_asynccnt 0x0" ::: "memory");
}

// WMMA A-matrix (16x32 bf16) fragment from row-major LDS tile [rows][ldk].
// lanes 0-15: row m, K = kb+0..15; lanes 16-31: K = kb+16..31.
__device__ __forceinline__ v16bf frag_a_lds(const unsigned short* base, int ldk,
                                            int row, int kb, int lane) {
  int half = lane >> 4;
  int m = lane & 15;
  const unsigned short* p0 = base + (row + m) * ldk + kb + half * 16;
  union { v16bf v; unsigned short u[16]; } t;
#pragma unroll
  for (int i = 0; i < 16; ++i) t.u[i] = p0[i];
  return t.v;
}

// WMMA B-matrix (32x16 bf16) fragment. LDS tile stored [N][K] row-major.
// lanes 0-15: col n, K = kb+0..15; lanes 16-31: K = kb+16..31.
__device__ __forceinline__ v16bf frag_b_lds(const unsigned short* base, int ldk,
                                            int col, int kb, int lane) {
  int half = lane >> 4;
  int n = lane & 15;
  const unsigned short* p = base + (col + n) * ldk + kb + half * 16;
  union { v16bf v; unsigned short u[16]; } t;
#pragma unroll
  for (int i = 0; i < 16; ++i) t.u[i] = p[i];
  return t.v;
}

// ---------------- Kernel 0: one-shot fp32 -> bf16 conversion ----------------
__global__ __launch_bounds__(256) void cvt_bf16_kernel(const float* __restrict__ src,
                                                       unsigned short* __restrict__ dst,
                                                       int n4) {
  int i = blockIdx.x * 256 + threadIdx.x;
  if (i < n4) {
    float4 x = ((const float4*)src)[i];
    ((ushort4*)dst)[i] = make_ushort4(f32_to_bf16(x.x), f32_to_bf16(x.y),
                                      f32_to_bf16(x.z), f32_to_bf16(x.w));
  }
}

// ---------------- Kernel 1: fused QKV GEMM (C = X * W^T), async double-buffered
// grid = (A/128, M/128, 3), block = 256. z==2 writes V transposed [b][a][t].
#define G_LDK 40

__global__ __launch_bounds__(256) void qkv_gemm_kernel(
    const unsigned short* __restrict__ Xb,
    const unsigned short* __restrict__ Wb,   // [3][A][E], order Q,K,V
    unsigned short* __restrict__ Qb, unsigned short* __restrict__ Kb,
    unsigned short* __restrict__ Vt) {
  __shared__ unsigned short sX[2][128 * G_LDK];   // 2 x 10,240B
  __shared__ unsigned short sW[2][128 * G_LDK];   // 2 x 10,240B

  const int z = blockIdx.z;
  const unsigned short* W = Wb + (size_t)z * A_DIM * E_DIM;

  const int mbase = blockIdx.y * 128;
  const int nbase = blockIdx.x * 128;
  const int tid  = threadIdx.x;
  const int lane = tid & 31;
  const int wv   = tid >> 5;          // wave 0..7
  const int wr   = (wv & 1) * 64;
  const int wc   = (wv >> 1) * 32;

  const int lrow = tid >> 1;          // 0..127
  const int lcol = (tid & 1) * 16;    // 0 or 16

  const unsigned short* gX = Xb + (size_t)(mbase + lrow) * E_DIM + lcol;
  const unsigned short* gW = W  + (size_t)(nbase + lrow) * E_DIM + lcol;
  const unsigned oX = lds_off(&sX[0][lrow * G_LDK + lcol]);
  const unsigned oW = lds_off(&sW[0][lrow * G_LDK + lcol]);
  const unsigned bufstep = 128 * G_LDK * 2;  // bytes between buffers

  auto issue = [&](int buf, int kb) {
    unsigned ox = oX + buf * bufstep;
    unsigned ow = oW + buf * bufstep;
    async_b128(ox,      gX + kb);
    async_b128(ox + 16, gX + kb + 8);
    async_b128(ow,      gW + kb);
    async_b128(ow + 16, gW + kb + 8);
  };

  v8f acc[4][2] = {};
  issue(0, 0);

  for (int kb = 0; kb < E_DIM; kb += 32) {
    int cur = (kb >> 5) & 1;
    wait_async0();
    __syncthreads();
    if (kb + 32 < E_DIM) issue(cur ^ 1, kb + 32);

    v16bf a[4], b[2];
#pragma unroll
    for (int i = 0; i < 4; ++i) a[i] = frag_a_lds(sX[cur], G_LDK, wr + i * 16, 0, lane);
#pragma unroll
    for (int j = 0; j < 2; ++j) b[j] = frag_b_lds(sW[cur], G_LDK, wc + j * 16, 0, lane);
#pragma unroll
    for (int i = 0; i < 4; ++i)
#pragma unroll
      for (int j = 0; j < 2; ++j)
        acc[i][j] = __builtin_amdgcn_wmma_f32_16x16x32_bf16(
            false, a[i], false, b[j], (short)0, acc[i][j], false, false);
  }

  const int half = lane >> 4;
  const int n = lane & 15;
#pragma unroll
  for (int i = 0; i < 4; ++i)
#pragma unroll
    for (int j = 0; j < 2; ++j)
#pragma unroll
      for (int v = 0; v < 8; ++v) {
        int gr = mbase + wr + i * 16 + v + half * 8;  // D-layout row
        int gc = nbase + wc + j * 16 + n;             // D-layout col
        unsigned short bv = f32_to_bf16(acc[i][j][v]);
        if (z == 0)      Qb[(size_t)gr * A_DIM + gc] = bv;
        else if (z == 1) Kb[(size_t)gr * A_DIM + gc] = bv;
        else {           // V stored transposed per batch: [b][a_col][t]
          int bb = gr >> 11, t = gr & (T_DIM - 1);
          Vt[((size_t)bb * A_DIM + gc) * T_DIM + t] = bv;
        }
      }
}

// ---------------- Kernel 2: causal flash attention, async double-buffered ----
// grid = (A/128, T/128, B), block = 256 (8 waves, each owning 16 query rows).
#define F_LDK 40
#define F_LV  72
#define F_LP  72

__global__ __launch_bounds__(256) void flash_attn_kernel(
    const unsigned short* __restrict__ Qb,
    const unsigned short* __restrict__ Kb,
    const unsigned short* __restrict__ Vt,
    float* __restrict__ out) {
  __shared__ unsigned short sQpool[2 * 128 * F_LDK];  // 20,480B; also holds P tiles
  __shared__ unsigned short sK[2][64 * F_LDK];        // 10,240B
  __shared__ unsigned short sVt[128 * F_LV];          // 18,432B  (total 48KB)

  const int cb    = blockIdx.x * 128;
  const int qt    = blockIdx.y;
  const int qbase = qt * 128;
  const int bz    = blockIdx.z;
  const int tid   = threadIdx.x;
  const int lane  = tid & 31;
  const int wv    = tid >> 5;          // 0..7
  const int half  = lane >> 4;
  const int n     = lane & 15;

  const unsigned short* Qg = Qb + ((size_t)bz * T_DIM + qbase) * A_DIM;
  const unsigned short* Kg = Kb + (size_t)bz * T_DIM * A_DIM;
  const unsigned short* Vg = Vt + ((size_t)bz * A_DIM + cb) * T_DIM;

  // async-issue addressing
  const int qrow = tid >> 1, qcol = (tid & 1) * 16;   // Q: 2 ops/thread
  const int krow = tid >> 2, kcol = (tid & 3) * 8;    // K: 1 op/thread
  const int vcol = tid >> 1, vk   = (tid & 1) * 32;   // V: 4 ops/thread

  const unsigned short* gQ = Qg + (size_t)qrow * A_DIM + qcol;
  const unsigned short* gK = Kg;  // + (jbase+krow)*A_DIM + kcol per tile
  const unsigned oQ = lds_off(&sQpool[qrow * F_LDK + qcol]);
  const unsigned oK = lds_off(&sK[0][krow * F_LDK + kcol]);
  const unsigned qstep = 128 * F_LDK * 2;
  const unsigned kstep = 64 * F_LDK * 2;

  auto issueQK = [&](int buf, int kk, int jbase) {
    unsigned oq = oQ + buf * qstep;
    async_b128(oq,      gQ + kk);
    async_b128(oq + 16, gQ + kk + 8);
    async_b128(oK + buf * kstep,
               gK + (size_t)(jbase + krow) * A_DIM + kk + kcol);
  };

  v8f o[8] = {};
  float m[8], l[8];
#pragma unroll
  for (int v = 0; v < 8; ++v) { m[v] = -3.0e38f; l[v] = 0.0f; }

  const float scale = 0.03125f;  // 1/sqrt(1024)
  const int jmax = 2 * qt + 1;   // last key tile overlapping the diagonal

  for (int j = 0; j <= jmax; ++j) {
    const int jbase = j * 64;
    __syncthreads();  // previous iteration's PV reads of sVt / P region done

    // V tile (already transposed in ws): sVt[c][k] = V[jbase+k][cb+c]
    {
      unsigned ov = lds_off(&sVt[vcol * F_LV + vk]);
      const unsigned short* gv = Vg + (size_t)vcol * T_DIM + jbase + vk;
      async_b128(ov,      gv);
      async_b128(ov + 16, gv + 8);
      async_b128(ov + 32, gv + 16);
      async_b128(ov + 48, gv + 24);
    }
    issueQK(0, 0, jbase);

    // S = Q * K^T, streaming the A-dim in chunks of 32
    v8f s[4] = {};
    for (int kk = 0; kk < A_DIM; kk += 32) {
      int cur = (kk >> 5) & 1;
      wait_async0();
      __syncthreads();
      if (kk + 32 < A_DIM) issueQK(cur ^ 1, kk + 32, jbase);
      v16bf a = frag_a_lds(sQpool + cur * 128 * F_LDK, F_LDK, wv * 16, 0, lane);
#pragma unroll
      for (int tc = 0; tc < 4; ++tc) {
        v16bf b = frag_b_lds(sK[cur], F_LDK, tc * 16, 0, lane);
        s[tc] = __builtin_amdgcn_wmma_f32_16x16x32_bf16(
            false, a, false, b, (short)0, s[tc], false, false);
      }
    }
    __syncthreads();  // all waves done with sQ before P aliases onto it

    // scale + causal mask (only the last two key tiles touch the diagonal band)
    if (j >= 2 * qt) {
#pragma unroll
      for (int tc = 0; tc < 4; ++tc)
#pragma unroll
        for (int v = 0; v < 8; ++v) {
          int kc = jbase + tc * 16 + n;
          int qr = qbase + wv * 16 + v + half * 8;
          float sv = s[tc][v] * scale;
          s[tc][v] = (kc > qr) ? -3.0e38f : sv;
        }
    } else {
#pragma unroll
      for (int tc = 0; tc < 4; ++tc)
#pragma unroll
        for (int v = 0; v < 8; ++v) s[tc][v] = s[tc][v] * scale;
    }

    // Online softmax: row (v + 8*half) spans 16 lanes of one VGPR.
    unsigned short* sPw = sQpool + wv * 16 * F_LP;  // P aliases the Q buffers
#pragma unroll
    for (int v = 0; v < 8; ++v) {
      float mx = fmaxf(fmaxf(s[0][v], s[1][v]), fmaxf(s[2][v], s[3][v]));
#pragma unroll
      for (int off = 1; off < 16; off <<= 1) mx = fmaxf(mx, __shfl_xor(mx, off, 16));
      float mn = fmaxf(m[v], mx);
      float al = __expf(m[v] - mn);
      float rs = 0.0f;
#pragma unroll
      for (int tc = 0; tc < 4; ++tc) {
        float p = __expf(s[tc][v] - mn);
        s[tc][v] = p;
        rs += p;
      }
#pragma unroll
      for (int off = 1; off < 16; off <<= 1) rs += __shfl_xor(rs, off, 16);
      l[v] = l[v] * al + rs;
      m[v] = mn;
#pragma unroll
      for (int ot = 0; ot < 8; ++ot) o[ot][v] *= al;
      int pr = v + half * 8;
#pragma unroll
      for (int tc = 0; tc < 4; ++tc)
        sPw[pr * F_LP + tc * 16 + n] = f32_to_bf16(s[tc][v]);
    }

    // O += P * V (P per-wave private; same-wave LDS ops are in order)
#pragma unroll
    for (int kb2 = 0; kb2 < 64; kb2 += 32) {
      v16bf pa = frag_a_lds(sPw, F_LP, 0, kb2, lane);
#pragma unroll
      for (int ot = 0; ot < 8; ++ot) {
        v16bf vb = frag_b_lds(sVt, F_LV, ot * 16, kb2, lane);
        o[ot] = __builtin_amdgcn_wmma_f32_16x16x32_bf16(
            false, pa, false, vb, (short)0, o[ot], false, false);
      }
    }
  }

  // Normalize + store fp32 (mirror reference's round-to-4-decimals)
  float inv[8];
#pragma unroll
  for (int v = 0; v < 8; ++v) inv[v] = 1.0f / l[v];
#pragma unroll
  for (int ot = 0; ot < 8; ++ot)
#pragma unroll
    for (int v = 0; v < 8; ++v) {
      int gr = qbase + wv * 16 + v + half * 8;
      int gc = cb + ot * 16 + n;
      float val = o[ot][v] * inv[v];
      out[((size_t)bz * T_DIM + gr) * A_DIM + gc] = rintf(val * 1.0e4f) * 1.0e-4f;
    }
}

extern "C" void kernel_launch(void* const* d_in, const int* in_sizes, int n_in,
                              void* d_out, int out_size, void* d_ws, size_t ws_size,
                              hipStream_t stream) {
  (void)in_sizes; (void)n_in; (void)out_size; (void)ws_size;
  const float* X  = (const float*)d_in[0];
  const float* Wk = (const float*)d_in[1];
  const float* Wq = (const float*)d_in[2];
  const float* Wv = (const float*)d_in[3];
  float* out = (float*)d_out;

  // ws: Qb(16MB) Kb(16MB) Vt(16MB) Xb(16MB) Wb[3](6MB) = 70MB
  const size_t QKV = (size_t)M_DIM * A_DIM;
  unsigned short* Qb = (unsigned short*)d_ws;
  unsigned short* Kb = Qb + QKV;
  unsigned short* Vt = Kb + QKV;
  unsigned short* Xb = Vt + QKV;
  unsigned short* Wb = Xb + QKV;  // [3][A][E], order Q,K,V

  const size_t WN = (size_t)A_DIM * E_DIM;
  cvt_bf16_kernel<<<(M_DIM * E_DIM / 4 + 255) / 256, 256, 0, stream>>>(X, Xb, M_DIM * E_DIM / 4);
  cvt_bf16_kernel<<<(WN / 4 + 255) / 256, 256, 0, stream>>>(Wq, Wb,          (int)(WN / 4));
  cvt_bf16_kernel<<<(WN / 4 + 255) / 256, 256, 0, stream>>>(Wk, Wb + WN,     (int)(WN / 4));
  cvt_bf16_kernel<<<(WN / 4 + 255) / 256, 256, 0, stream>>>(Wv, Wb + 2 * WN, (int)(WN / 4));

  dim3 g1(A_DIM / 128, M_DIM / 128, 3);
  qkv_gemm_kernel<<<g1, 256, 0, stream>>>(Xb, Wb, Qb, Kb, Vt);

  dim3 g2(A_DIM / 128, T_DIM / 128, B_DIM);
  flash_attn_kernel<<<g2, 256, 0, stream>>>(Qb, Kb, Vt, out);
}
